// MambaSSM_14242111553797
// MI455X (gfx1250) — matrix operand order repaired and checked by reading, not compiled, not run on hardware
//
#include <hip/hip_runtime.h>

typedef __attribute__((ext_vector_type(16))) __bf16 v16bf;
typedef __attribute__((ext_vector_type(8)))  __bf16 v8bf;
typedef __attribute__((ext_vector_type(8)))  float  v8f;

#define D_MODEL 1024
#define D_STATE 16
#define D_CONV  4
#define D_INNER 2048
#define BB 2
#define LL 2048
#define ROWS (BB * LL)   // 4096 token rows

// Async global->LDS (CDNA5 GLOBAL_LOAD_ASYNC_TO_LDS_*, ASYNCcnt) if this
// toolchain exposes the builtin; otherwise fall back to load+ds_store staging.
#if defined(__has_builtin)
#if __has_builtin(__builtin_amdgcn_global_load_async_to_lds_b128)
#define HAVE_ASYNC_LDS 1
#endif
#endif

// The builtin's pointer params are native 4xint vector pointers
// (diagnostic: "__attribute__((__vector_size__(4 * sizeof(int)))) int *").
typedef int v4i_n __attribute__((vector_size(16)));
#if defined(__HIP_DEVICE_COMPILE__)
typedef __attribute__((address_space(3))) v4i_n lds_v4i;
#define GLOBAL_V4I(p) ((v4i_n*)(p))
#define LDS_V4I(p)    ((lds_v4i*)(p))
#endif

__device__ __forceinline__ void wait_async0() {
#if defined(__has_builtin) && __has_builtin(__builtin_amdgcn_s_wait_asynccnt)
    __builtin_amdgcn_s_wait_asynccnt(0);
#else
    asm volatile("s_wait_asynccnt 0x0" ::: "memory");
#endif
}

// ---------------------------------------------------------------------------
// f32 -> bf16 elementwise convert
// ---------------------------------------------------------------------------
__global__ void mamba_cvt_bf16(const float* __restrict__ in, __bf16* __restrict__ out, int n) {
    int i = blockIdx.x * blockDim.x + threadIdx.x;
    if (i < n) out[i] = (__bf16)in[i];
}

// in: K x N (f32, row-major) ; out: N x K (bf16) so GEMM B-tiles are k-contiguous
__global__ void mamba_cvt_bf16_T(const float* __restrict__ in, __bf16* __restrict__ out, int K, int N) {
    int i = blockIdx.x * blockDim.x + threadIdx.x;
    if (i < K * N) {
        int k = i % K;
        int n = i / K;
        out[(size_t)n * K + k] = (__bf16)in[(size_t)k * N + n];
    }
}

// ---------------------------------------------------------------------------
// bf16 WMMA GEMM: C[M,N] (f32) = A[M,K] (bf16 row-major) * Bt[N,K]^T (bf16)
// block = 128 threads (4 wave32), block tile 128(M) x 64(N), K-step 32.
// Each wave owns a 32x64 slab: 2x4 tiles of 16x16, v_wmma_f32_16x16x32_bf16.
// Tile staging uses async global->LDS copies when available.
// ---------------------------------------------------------------------------
__global__ __launch_bounds__(128)
void mamba_gemm_bf16_wmma(const __bf16* __restrict__ A, const __bf16* __restrict__ Bt,
                          float* __restrict__ C, int M, int N, int K) {
    __shared__ __align__(16) __bf16 Alds[128 * 32];   // 8 KB
    __shared__ __align__(16) __bf16 Blds[64 * 32];    // 4 KB
    const int tid  = threadIdx.x;
    const int lane = tid & 31;
    const int wave = tid >> 5;
    const int m0g  = blockIdx.y * 128;
    const int n0g  = blockIdx.x * 64;
    const int g    = lane >> 4;   // lane half (0/1)
    const int r    = lane & 15;

    v8f acc[2][4];
#pragma unroll
    for (int mt = 0; mt < 2; ++mt)
#pragma unroll
        for (int nt = 0; nt < 4; ++nt) {
            v8f z = {0.f, 0.f, 0.f, 0.f, 0.f, 0.f, 0.f, 0.f};
            acc[mt][nt] = z;
        }

    for (int k0 = 0; k0 < K; k0 += 32) {
        const __bf16* srcA = A + (size_t)(m0g + tid) * K + k0;      // 64B per thread
        const int nB  = tid >> 1;
        const int khB = (tid & 1) * 16;
        const __bf16* srcB = Bt + (size_t)(n0g + nB) * K + k0 + khB; // 32B per thread
        __bf16* dstA = Alds + tid * 32;
        __bf16* dstB = Blds + nB * 32 + khB;

#if defined(__HIP_DEVICE_COMPILE__) && HAVE_ASYNC_LDS
        // DMA-style staging: memory -> LDS without VGPR round-trip (ASYNCcnt).
        __builtin_amdgcn_global_load_async_to_lds_b128(GLOBAL_V4I(srcA), LDS_V4I(dstA), 0, 0);
        __builtin_amdgcn_global_load_async_to_lds_b128(GLOBAL_V4I(srcA), LDS_V4I(dstA), 16, 0);
        __builtin_amdgcn_global_load_async_to_lds_b128(GLOBAL_V4I(srcA), LDS_V4I(dstA), 32, 0);
        __builtin_amdgcn_global_load_async_to_lds_b128(GLOBAL_V4I(srcA), LDS_V4I(dstA), 48, 0);
        __builtin_amdgcn_global_load_async_to_lds_b128(GLOBAL_V4I(srcB), LDS_V4I(dstB), 0, 0);
        __builtin_amdgcn_global_load_async_to_lds_b128(GLOBAL_V4I(srcB), LDS_V4I(dstB), 16, 0);
        if (k0 + 32 < K)
            __builtin_prefetch(srcA + 32, 0, 1);
        wait_async0();
#else
        {
            const uint4* sa = (const uint4*)srcA;
            uint4* da = (uint4*)dstA;
            da[0] = sa[0]; da[1] = sa[1]; da[2] = sa[2]; da[3] = sa[3];
            const uint4* sb = (const uint4*)srcB;
            uint4* db = (uint4*)dstB;
            db[0] = sb[0]; db[1] = sb[1];
            if (k0 + 32 < K)
                __builtin_prefetch(srcA + 32, 0, 1);
        }
#endif
        __syncthreads();

#pragma unroll
        for (int mt = 0; mt < 2; ++mt) {
            // 16-bit A 16x32 layout: lane group g holds K = {g*8..g*8+7, 16+g*8..16+g*8+7}
            const __bf16* ap = Alds + ((wave * 32) + mt * 16 + r) * 32;
            v8bf alo = *(const v8bf*)(ap + g * 8);
            v8bf ahi = *(const v8bf*)(ap + 16 + g * 8);
            v16bf a = __builtin_shufflevector(alo, ahi,
                0, 1, 2, 3, 4, 5, 6, 7, 8, 9, 10, 11, 12, 13, 14, 15);
#pragma unroll
            for (int nt = 0; nt < 4; ++nt) {
                // B 32x16 layout: lane group g holds K = {g*8..g*8+7, 16+g*8..16+g*8+7}, N = lane&15
                const __bf16* bp = Blds + (nt * 16 + r) * 32;
                v8bf blo = *(const v8bf*)(bp + g * 8);
                v8bf bhi = *(const v8bf*)(bp + 16 + g * 8);
                v16bf bfr = __builtin_shufflevector(blo, bhi,
                    0, 1, 2, 3, 4, 5, 6, 7, 8, 9, 10, 11, 12, 13, 14, 15);
                acc[mt][nt] = __builtin_amdgcn_wmma_f32_16x16x32_bf16(
                    false, a, false, bfr, (short)0, acc[mt][nt], false, false);
            }
        }
        __syncthreads();
    }

    // C/D 16x16 f32 layout: VGPR i, lane l -> row = (l>>4)*8 + i, col = l&15
#pragma unroll
    for (int mt = 0; mt < 2; ++mt)
#pragma unroll
        for (int nt = 0; nt < 4; ++nt)
#pragma unroll
            for (int i = 0; i < 8; ++i) {
                int row = m0g + wave * 32 + mt * 16 + g * 8 + i;
                int col = n0g + nt * 16 + r;
                C[(size_t)row * N + col] = acc[mt][nt][i];
            }
}

// ---------------------------------------------------------------------------
// Depthwise causal conv(4) + bias + SiLU over xz's x-half -> xs (f32)
// xz layout: (b, l, 2*D_INNER), x-half = [0, D_INNER)
// ---------------------------------------------------------------------------
__global__ void mamba_conv_silu(const float* __restrict__ xz, const float* __restrict__ cw,
                                const float* __restrict__ cb, float* __restrict__ xs) {
    int idx = blockIdx.x * blockDim.x + threadIdx.x;
    if (idx >= BB * LL * D_INNER) return;
    int d = idx & (D_INNER - 1);
    int l = (idx >> 11) & (LL - 1);
    int b = idx >> 22;
    float acc = cb[d];
#pragma unroll
    for (int k = 0; k < D_CONV; ++k) {
        int ll = l - (D_CONV - 1) + k;
        if (ll >= 0)
            acc += xz[((size_t)(b * LL + ll)) * (2 * D_INNER) + d] * cw[d * D_CONV + k];
    }
    xs[idx] = acc * (1.f / (1.f + __expf(-acc)));
}

// ---------------------------------------------------------------------------
// x_proj: bcd = xs_row(2048) . W(2048x33)  ->  B(16), C(16), dt_raw(1)
// one block per token row, threads 0..32 each compute one output column
// ---------------------------------------------------------------------------
__global__ __launch_bounds__(64)
void mamba_xproj(const float* __restrict__ xs, const float* __restrict__ w,
                 float* __restrict__ Bm, float* __restrict__ Cc, float* __restrict__ dtraw) {
    int row = blockIdx.x;
    int p = threadIdx.x;
    if (p >= 33) return;
    const float* xr = xs + (size_t)row * D_INNER;
    float acc = 0.f;
    for (int k = 0; k < D_INNER; ++k)
        acc += xr[k] * w[k * 33 + p];
    if (p < 16)      Bm[row * 16 + p] = acc;
    else if (p < 32) Cc[row * 16 + (p - 16)] = acc;
    else             dtraw[row] = acc;
}

// ---------------------------------------------------------------------------
// Sequential selective scan over L, fused with dt softplus, dA=exp, D skip,
// and silu(z) gate. One thread per (b, d) channel; h[16] in VGPRs.
// B/C/dt chunks staged in LDS. Writes bf16 for the final out_proj GEMM.
// ---------------------------------------------------------------------------
#define SCAN_CH 64
__global__ __launch_bounds__(256)
void mamba_scan(const float* __restrict__ xz, const float* __restrict__ xs,
                const float* __restrict__ Bm, const float* __restrict__ Cc,
                const float* __restrict__ dtraw,
                const float* __restrict__ dt_w, const float* __restrict__ dt_b,
                const float* __restrict__ A_log, const float* __restrict__ Dp,
                __bf16* __restrict__ yg) {
    __shared__ float sB[SCAN_CH * 16];
    __shared__ float sC[SCAN_CH * 16];
    __shared__ float sdt[SCAN_CH];
    const int tid = threadIdx.x;
    const int b = blockIdx.x >> 3;
    const int d = ((blockIdx.x & 7) << 8) + tid;
    const float dw = dt_w[d], db = dt_b[d], Dv = Dp[d];
    float negA[16], h[16];
#pragma unroll
    for (int n = 0; n < 16; ++n) { negA[n] = -__expf(A_log[d * 16 + n]); h[n] = 0.f; }

    for (int lc = 0; lc < LL; lc += SCAN_CH) {
        __syncthreads();
        const size_t base = ((size_t)b * LL + lc) * 16;
#pragma unroll
        for (int j = 0; j < 4; ++j) {
            sB[tid + j * 256] = Bm[base + tid + j * 256];
            sC[tid + j * 256] = Cc[base + tid + j * 256];
        }
        if (tid < SCAN_CH) sdt[tid] = dtraw[(size_t)b * LL + lc + tid];
        __syncthreads();

        for (int j = 0; j < SCAN_CH; ++j) {
            const size_t rl = (size_t)b * LL + lc + j;
            float xarg = sdt[j] * dw + db;
            float dt = (xarg > 20.f) ? xarg : log1pf(__expf(xarg));
            float xv = xs[rl * D_INNER + d];
            float zv = xz[rl * (2 * D_INNER) + D_INNER + d];
            float dtx = dt * xv;
            float y = 0.f;
#pragma unroll
            for (int n = 0; n < 16; ++n) {
                float dA = __expf(dt * negA[n]);
                h[n] = dA * h[n] + dtx * sB[j * 16 + n];
                y += h[n] * sC[j * 16 + n];
            }
            y += Dv * xv;
            float gate = zv * (1.f / (1.f + __expf(-zv)));
            yg[rl * D_INNER + d] = (__bf16)(y * gate);
        }
    }
}

// ---------------------------------------------------------------------------
// launch
// ---------------------------------------------------------------------------
extern "C" void kernel_launch(void* const* d_in, const int* in_sizes, int n_in,
                              void* d_out, int out_size, void* d_ws, size_t ws_size,
                              hipStream_t stream) {
    const float* x          = (const float*)d_in[0];
    const float* in_proj_w  = (const float*)d_in[1];
    const float* conv_w     = (const float*)d_in[2];
    const float* conv_b     = (const float*)d_in[3];
    const float* x_proj_w   = (const float*)d_in[4];
    const float* dt_w       = (const float*)d_in[5];
    const float* dt_b       = (const float*)d_in[6];
    const float* A_log      = (const float*)d_in[7];
    const float* Dp         = (const float*)d_in[8];
    const float* out_proj_w = (const float*)d_in[9];
    (void)in_sizes; (void)n_in; (void)out_size; (void)ws_size;

    char* wsp = (char*)d_ws;
    auto alloc = [&](size_t bytes) -> char* {
        char* p = wsp;
        wsp += (bytes + 255) & ~(size_t)255;
        return p;
    };
    __bf16* Xbf   = (__bf16*)alloc((size_t)ROWS * D_MODEL * 2);          // 8 MB
    __bf16* W1T   = (__bf16*)alloc((size_t)(2 * D_INNER) * D_MODEL * 2); // 8 MB  (N=4096 x K=1024)
    __bf16* W2T   = (__bf16*)alloc((size_t)D_MODEL * D_INNER * 2);       // 4 MB  (N=1024 x K=2048)
    float*  xz    = (float*) alloc((size_t)ROWS * 2 * D_INNER * 4);      // 64 MB
    float*  xs    = (float*) alloc((size_t)ROWS * D_INNER * 4);          // 32 MB
    float*  Bm    = (float*) alloc((size_t)ROWS * 16 * 4);
    float*  Cc    = (float*) alloc((size_t)ROWS * 16 * 4);
    float*  dtraw = (float*) alloc((size_t)ROWS * 4);
    __bf16* yg    = (__bf16*)alloc((size_t)ROWS * D_INNER * 2);          // 16 MB

    // 1) bf16 conversions
    mamba_cvt_bf16<<<(ROWS * D_MODEL) / 256, 256, 0, stream>>>(x, Xbf, ROWS * D_MODEL);
    mamba_cvt_bf16_T<<<(D_MODEL * 2 * D_INNER) / 256, 256, 0, stream>>>(in_proj_w, W1T, D_MODEL, 2 * D_INNER);
    mamba_cvt_bf16_T<<<(D_INNER * D_MODEL) / 256, 256, 0, stream>>>(out_proj_w, W2T, D_INNER, D_MODEL);

    // 2) in_proj GEMM: (4096x1024)x(1024x4096) -> xz
    dim3 g1((2 * D_INNER) / 64, ROWS / 128);
    mamba_gemm_bf16_wmma<<<g1, 128, 0, stream>>>(Xbf, W1T, xz, ROWS, 2 * D_INNER, D_MODEL);

    // 3) depthwise conv + SiLU
    mamba_conv_silu<<<(BB * LL * D_INNER) / 256, 256, 0, stream>>>(xz, conv_w, conv_b, xs);

    // 4) x_proj (B, C, dt_raw)
    mamba_xproj<<<ROWS, 64, 0, stream>>>(xs, x_proj_w, Bm, Cc, dtraw);

    // 5) selective scan (fused softplus/exp/D-skip/silu(z) gate) -> bf16 yg
    mamba_scan<<<BB * (D_INNER / 256), 256, 0, stream>>>(xz, xs, Bm, Cc, dtraw,
                                                         dt_w, dt_b, A_log, Dp, yg);

    // 6) out_proj GEMM: (4096x2048)x(2048x1024) -> d_out (f32)
    dim3 g2(D_MODEL / 64, ROWS / 128);
    mamba_gemm_bf16_wmma<<<g2, 128, 0, stream>>>(yg, W2T, (float*)d_out, ROWS, D_MODEL, D_INNER);
}
